// GraphAttentionLayer_73177652789832
// MI455X (gfx1250) — matrix operand order repaired and checked by reading, not compiled, not run on hardware
//
#include <hip/hip_runtime.h>

typedef __bf16 bf16;
typedef __attribute__((ext_vector_type(16))) __bf16 v16bf;
typedef __attribute__((ext_vector_type(8)))  __bf16 v8bf;
typedef __attribute__((ext_vector_type(8)))  float  v8f;
typedef __attribute__((ext_vector_type(4)))  float  v4f;

static constexpr int Bsz = 2;
static constexpr int Nn  = 2048;
static constexpr int FIN = 512;
static constexpr int Hh  = 8;
static constexpr int DHh = 64;
static constexpr int HD  = 512;     // H*DH
static constexpr float SLOPE = 0.2f;

static constexpr int LDS_PAD  = 8;          // bf16 elems: row stride 144 B (36 banks)
static constexpr int LDS_ROW  = 64 + LDS_PAD;
static constexpr int JSTEP    = 64;         // j per pipeline stage
static constexpr int NSTAGE   = Nn / JSTEP; // 32

__device__ __forceinline__ v16bf cat16(v8bf lo, v8bf hi) {
    return __builtin_shufflevector(lo, hi, 0, 1, 2, 3, 4, 5, 6, 7,
                                           8, 9, 10, 11, 12, 13, 14, 15);
}

// ---------------- K0a: f (fp32) -> bf16 ----------------
__global__ void gat_cvt_f(const float* __restrict__ f, bf16* __restrict__ fb) {
    int t = blockIdx.x * blockDim.x + threadIdx.x;
    if (t < Bsz * Nn * FIN) fb[t] = (bf16)f[t];
}

// ---------------- K0b: W (fp32, FIN x HD) -> Wt bf16 (HD x FIN) -------------
__global__ void gat_trs_w(const float* __restrict__ W, bf16* __restrict__ Wt) {
    int t = blockIdx.x * blockDim.x + threadIdx.x;
    if (t < FIN * HD) {
        int n = t / FIN;                       // cout
        int k = t % FIN;                       // cin
        Wt[t] = (bf16)W[k * HD + n];           // coalesced write
    }
}

// ---------------- K1: g = f @ W, stored transposed g_t[b][h][d][n] (bf16) ---
// True ping-pong (K unrolled by 2): each stage loads straight into the
// registers its WMMAs consume -> no rotation movs, no mov->wmma nops.
__global__ __launch_bounds__(256) void gat_gemm1(const bf16* __restrict__ fb,
                                                 const bf16* __restrict__ Wt,
                                                 bf16* __restrict__ gt) {
    const int lane   = threadIdx.x & 31;
    const int wave   = threadIdx.x >> 5;
    const bool laneHi = lane >= 16;
    const int lm     = lane & 15;
    const int i0     = blockIdx.x * 128 + wave * 16;   // global row (b*N+n)
    const int c0     = blockIdx.y * 64;                // output col (h*64+d)

    v8f acc[4] = {};
    const bf16* fptr = fb + (i0 + lm) * FIN;
    const int ka0 = laneHi ? 16 : 0;   // A: K half (contiguous)
    const int kb0 = laneHi ? 16 : 0;   // B: K half

    auto loadA = [&](int k0) {
        v8bf alo = *(const v8bf*)(fptr + k0 + ka0);
        v8bf ahi = *(const v8bf*)(fptr + k0 + ka0 + 8);
        return cat16(alo, ahi);
    };
    auto loadB = [&](int t, int k0) {
        return *(const v16bf*)(Wt + (c0 + t * 16 + lm) * FIN + k0 + kb0);
    };

    v16bf aP[2];
    v16bf bP[2][4];
    aP[0] = loadA(0);
    #pragma unroll
    for (int t = 0; t < 4; ++t) bP[0][t] = loadB(t, 0);

    #pragma unroll 1
    for (int k0 = 0; k0 < FIN; k0 += 64) {
        // stage k0+32 into slot 1 while slot 0 computes
        aP[1] = loadA(k0 + 32);
        #pragma unroll
        for (int t = 0; t < 4; ++t) bP[1][t] = loadB(t, k0 + 32);
        __builtin_prefetch(fptr + k0 + 128, 0, 3);
        #pragma unroll
        for (int t = 0; t < 4; ++t)
            acc[t] = __builtin_amdgcn_wmma_f32_16x16x32_bf16(
                false, aP[0], false, bP[0][t], (short)0, acc[t], false, false);

        const int kn = k0 + 64;
        if (kn < FIN) {                 // stage kn into slot 0 under slot 1
            aP[0] = loadA(kn);
            #pragma unroll
            for (int t = 0; t < 4; ++t) bP[0][t] = loadB(t, kn);
        }
        #pragma unroll
        for (int t = 0; t < 4; ++t)
            acc[t] = __builtin_amdgcn_wmma_f32_16x16x32_bf16(
                false, aP[1], false, bP[1][t], (short)0, acc[t], false, false);
    }

    // D tile: VGPR r, lane<16 -> (M=r, N=lm); lane>=16 -> (M=r+8, N=lm).
    // Per lane/tile the 8 acc values are contiguous in n -> one b16x8 store.
    const int b_ = i0 / Nn;
    const int n0 = (i0 % Nn) + (laneHi ? 8 : 0);
    #pragma unroll
    for (int t = 0; t < 4; ++t) {
        const int c = c0 + t * 16 + lm;
        const int h = c >> 6, d = c & 63;
        v8bf o;
        #pragma unroll
        for (int r = 0; r < 8; ++r) o[r] = (bf16)acc[t][r];
        *(v8bf*)(gt + ((size_t)(b_ * Hh + h) * DHh + d) * Nn + n0) = o;
    }
}

// ---------------- K2: scores s_l, s_r per (b,h,n) from bf16 g_t -------------
__global__ void gat_scores(const bf16* __restrict__ gt, const float* __restrict__ aw,
                           float* __restrict__ sl, float* __restrict__ sr) {
    int t = blockIdx.x * blockDim.x + threadIdx.x;      // over B*H*N
    if (t >= Bsz * Hh * Nn) return;
    int n = t % Nn, bh = t / Nn;
    const bf16* col = gt + (size_t)bh * DHh * Nn + n;
    float al = 0.f, ar = 0.f;
    for (int d = 0; d < DHh; ++d) {
        float g = (float)col[(size_t)d * Nn];           // coalesced across n
        al += g * aw[d];
        ar += g * aw[DHh + d];
    }
    sl[t] = al; sr[t] = ar;
}

// ---------------- K3: per-row softmax stats m_i, Z_i (LDS staged) -----------
__global__ __launch_bounds__(256) void gat_stats(const float* __restrict__ sl,
                                                 const float* __restrict__ sr,
                                                 float* __restrict__ mx,
                                                 float* __restrict__ zz) {
    __shared__ float srs[Nn];
    const int bh = blockIdx.y;
    const int i  = blockIdx.x * 256 + threadIdx.x;
    for (int j = threadIdx.x; j < Nn; j += 256) srs[j] = sr[bh * Nn + j];
    __syncthreads();
    const float s = sl[bh * Nn + i];
    float m = -3.4e38f;
    for (int j = 0; j < Nn; ++j) {
        float e = s + srs[j];
        e = e >= 0.f ? e : SLOPE * e;
        m = fmaxf(m, e);
    }
    float z = 0.f;
    for (int j = 0; j < Nn; ++j) {
        float e = s + srs[j];
        e = e >= 0.f ? e : SLOPE * e;
        z += __expf(e - m);
    }
    mx[bh * Nn + i] = m;
    zz[bh * Nn + i] = z;
}

// ---------------- K4: out[d,i] = (1/Z_i) * sum_j exp(lrelu(sl_i+sr_j)-m_i)*g_t[d,j]
// A tiles (g_t[d,j-chunk]) are identical across all 8 waves of the block, so
// each 64x64 bf16 stage is DMA'd ONCE per block into LDS with
// global_load_async_to_lds_b128 (ASYNCcnt-tracked, double-buffered), then read
// back with ds_load_b128. B = exp-weights generated in-register; one B frag
// feeds 4 WMMAs.
__global__ __launch_bounds__(256) void gat_attn(const bf16* __restrict__ gt,
                                                const float* __restrict__ sl,
                                                const float* __restrict__ sr,
                                                const float* __restrict__ mx,
                                                const float* __restrict__ zz,
                                                float* __restrict__ out) {
    __shared__ bf16 smem[2][64 * LDS_ROW];              // 2 x 9 KiB stages

    const int lane   = threadIdx.x & 31;
    const int wave   = threadIdx.x >> 5;
    const bool laneHi = lane >= 16;
    const int lm     = lane & 15;
    const int bh     = blockIdx.y;
    const int b_ = bh / Hh, h = bh % Hh;
    const int i  = blockIdx.x * 128 + wave * 16 + lm;   // this lane's column i

    const float sli = sl[bh * Nn + i];
    const float mi  = mx[bh * Nn + i];
    v8f acc[4] = {};
    const int ka0 = laneHi ? 16 : 0;
    const int kb0 = laneHi ? 16 : 0;
    const bf16* gbase = gt + (size_t)bh * DHh * Nn;

    // cooperative copy mapping: 256 threads x 32 B cover one 64x64 bf16 stage
    const int crow = threadIdx.x >> 2;                  // d row 0..63
    const int cseg = (threadIdx.x & 3) * 16;            // elem offset 0/16/32/48

    auto issue_stage = [&](int j0, int bufIdx) {
        const bf16* gsrc = gbase + (size_t)crow * Nn + j0 + cseg;
        unsigned lds0 =
            (unsigned)(uintptr_t)(&smem[bufIdx][crow * LDS_ROW + cseg]);
        unsigned long long ga0 = (unsigned long long)(uintptr_t)gsrc;
        asm volatile("global_load_async_to_lds_b128 %0, %1, off"
                     :: "v"(lds0),       "v"(ga0)        : "memory");
        asm volatile("global_load_async_to_lds_b128 %0, %1, off"
                     :: "v"(lds0 + 16u), "v"(ga0 + 16ull) : "memory");
    };

    issue_stage(0, 0);
    for (int s = 0; s < NSTAGE; ++s) {
        const int j0 = s * JSTEP;
        if (s + 1 < NSTAGE) {
            issue_stage(j0 + JSTEP, (s + 1) & 1);
            asm volatile("s_wait_asynccnt 0x2" ::: "memory");  // stage s done
        } else {
            asm volatile("s_wait_asynccnt 0x0" ::: "memory");
        }
        __syncthreads();                                 // all waves' DMAs done

        #pragma unroll
        for (int u = 0; u < 2; ++u) {
            const int jj = j0 + u * 32;
            // B fragment: exp-weights for K = jj+kb0 .. +15 (this lane's half)
            v16bf bw;
            const float* srp = sr + bh * Nn + jj + kb0;
            __builtin_prefetch(srp + 64, 0, 3);
            #pragma unroll
            for (int q = 0; q < 4; ++q) {
                v4f sv = *(const v4f*)(srp + q * 4);
                #pragma unroll
                for (int e = 0; e < 4; ++e) {
                    float ee = sli + sv[e];
                    ee = ee >= 0.f ? ee : SLOPE * ee;
                    bw[q * 4 + e] = (bf16)__expf(ee - mi);
                }
            }
            #pragma unroll
            for (int t = 0; t < 4; ++t) {
                const int arow = (t * 16 + lm) * LDS_ROW + u * 32 + ka0;
                v8bf alo = *(const v8bf*)(&smem[s & 1][arow]);
                v8bf ahi = *(const v8bf*)(&smem[s & 1][arow + 8]);
                v16bf a = cat16(alo, ahi);
                acc[t] = __builtin_amdgcn_wmma_f32_16x16x32_bf16(
                    false, a, false, bw, (short)0, acc[t], false, false);
            }
        }
        __syncthreads();                // reads done before buffer is reused
    }

    const float rz = 1.0f / zz[bh * Nn + i];
    float* op = out + ((size_t)b_ * Nn + i) * HD + h * DHh;
    #pragma unroll
    for (int t = 0; t < 4; ++t) {
        const int d0 = t * 16 + (laneHi ? 8 : 0);       // rows contiguous in d
        v4f o0, o1;
        #pragma unroll
        for (int e = 0; e < 4; ++e) { o0[e] = acc[t][e] * rz; o1[e] = acc[t][4 + e] * rz; }
        *(v4f*)(op + d0)     = o0;
        *(v4f*)(op + d0 + 4) = o1;
    }
}

// ---------------- launch ----------------------------------------------------
extern "C" void kernel_launch(void* const* d_in, const int* in_sizes, int n_in,
                              void* d_out, int out_size, void* d_ws, size_t ws_size,
                              hipStream_t stream) {
    const float* f  = (const float*)d_in[0];
    // d_in[1] = adj_mat (all-ones, unused by the reference)
    const float* W  = (const float*)d_in[2];
    const float* aw = (const float*)d_in[3];
    float* out = (float*)d_out;

    char* ws = (char*)d_ws;
    bf16*  fb = (bf16*)(ws);                         //  4 MiB  f in bf16
    bf16*  Wt = (bf16*)(ws + 4194304);               //  512 KiB W^T bf16
    bf16*  gt = (bf16*)(ws + 4718592);               //  4 MiB  g_t[b][h][d][n]
    float* sl = (float*)(ws + 8912896);              //  128 KiB
    float* sr = (float*)(ws + 9043968);              //  128 KiB
    float* mx = (float*)(ws + 9175040);              //  128 KiB
    float* zz = (float*)(ws + 9306112);              //  128 KiB

    gat_cvt_f <<<(Bsz * Nn * FIN) / 256, 256, 0, stream>>>(f, fb);
    gat_trs_w <<<(FIN * HD) / 256,       256, 0, stream>>>(W, Wt);
    gat_gemm1 <<<dim3(Bsz * Nn / 128, HD / 64), 256, 0, stream>>>(fb, Wt, gt);
    gat_scores<<<(Bsz * Hh * Nn) / 256,  256, 0, stream>>>(gt, aw, sl, sr);
    gat_stats <<<dim3(Nn / 256, Bsz * Hh), 256, 0, stream>>>(sl, sr, mx, zz);
    gat_attn  <<<dim3(Nn / 128, Bsz * Hh), 256, 0, stream>>>(gt, sl, sr, mx, zz, out);
}
